// DynamicSparseMoE_30623116821367
// MI455X (gfx1250) — compile-verified, matrix operand order checked by reading, not run on hardware
//
#include <hip/hip_runtime.h>
#include <stddef.h>

// ---------------- problem constants ----------------
#define NUM_EXPERTS 8
#define T_TOK       8192
#define D_IN        2048
#define D_OUT       2048

// ---------------- tiling ----------------
#define BM 128            // token rows per block tile
#define BN 64             // output cols per block tile
#define BK 32             // K-step (one bf16 WMMA K)
#define LDK 72            // padded LDS row stride in bf16 elems (144 B, 16B-aligned, conflict-free)
#define MT_MAX 72         // max padded M tiles: T/BM + E = 64 + 8
#define PERM_CAP (T_TOK + NUM_EXPERTS * BM)   // 9216 padded slots

typedef __bf16 v16bf __attribute__((ext_vector_type(16)));
typedef __bf16 v8bf  __attribute__((ext_vector_type(8)));
typedef float  v8f   __attribute__((ext_vector_type(8)));

union FragU { v16bf v; v8bf h[2]; };

// workspace layout (ints): counts[8] | offsets[8] | cursor[8] | tileExpert[96] | perm[PERM_CAP]
#define WS_COUNTS 0
#define WS_OFFS   8
#define WS_CURS   16
#define WS_TILEE  24
#define WS_PERM   128

// ---------------- setup kernels ----------------
__global__ void moe_init_kernel(int* ws) {
    int i = blockIdx.x * blockDim.x + threadIdx.x;
    if (i < NUM_EXPERTS) ws[WS_COUNTS + i] = 0;
    if (i < 96)          ws[WS_TILEE + i]  = -1;
    if (i < PERM_CAP)    ws[WS_PERM + i]   = -1;
}

__global__ void moe_count_kernel(const int* __restrict__ exp_ids, int* ws) {
    int t = blockIdx.x * blockDim.x + threadIdx.x;
    if (t < T_TOK) atomicAdd(&ws[WS_COUNTS + exp_ids[t]], 1);
}

__global__ void moe_scan_kernel(int* ws) {
    if (threadIdx.x != 0 || blockIdx.x != 0) return;
    int base = 0;
    for (int e = 0; e < NUM_EXPERTS; ++e) {
        ws[WS_OFFS + e] = base;
        ws[WS_CURS + e] = base;
        int cnt = ws[WS_COUNTS + e];
        int nt  = (cnt + BM - 1) / BM;
        for (int i = 0; i < nt; ++i) ws[WS_TILEE + base / BM + i] = e;
        base += nt * BM;
    }
}

__global__ void moe_scatter_kernel(const int* __restrict__ exp_ids, int* ws) {
    int t = blockIdx.x * blockDim.x + threadIdx.x;
    if (t < T_TOK) {
        int e = exp_ids[t];
        int slot = atomicAdd(&ws[WS_CURS + e], 1);
        ws[WS_PERM + slot] = t;
    }
}

// ---------------- grouped GEMM kernel ----------------
// 256 threads = 8 wave32 waves, arranged 4(M) x 2(N); each wave owns 32x32.
// fp32 inputs split into hi/lo bf16; D = Ahi*Bhi + Ahi*Blo + Alo*Bhi (fp32 acc).
// Register double-buffered: global loads for step k+1 overlap WMMAs of step k.
__launch_bounds__(256)
__global__ void moe_gemm_kernel(const float* __restrict__ tokens,
                                const float* __restrict__ weight,
                                const int*   __restrict__ ws,
                                float*       __restrict__ out) {
    __shared__ __align__(16) __bf16 AsHi[BM * LDK];
    __shared__ __align__(16) __bf16 AsLo[BM * LDK];
    __shared__ __align__(16) __bf16 BsHi[BN * LDK];
    __shared__ __align__(16) __bf16 BsLo[BN * LDK];

    const int mt = blockIdx.x;
    const int e  = ws[WS_TILEE + mt];
    if (e < 0) return;                       // unused padded tile
    const int n0 = blockIdx.y * BN;

    const int tid  = threadIdx.x;
    const int lane = tid & 31;
    const int w    = tid >> 5;
    const int wm   = w >> 1;                 // 0..3 -> row group of 32
    const int wn   = w & 1;                  // 0..1 -> col group of 32

    // A-tile loader mapping: 128 rows x 32 cols, float4 granularity
    const int arow = tid >> 3;               // 0..31 (x4 reps -> 128 rows)
    const int ac4  = (tid & 7) * 4;          // col base (floats)
    int tokRow[4];
#pragma unroll
    for (int i = 0; i < 4; ++i) tokRow[i] = ws[WS_PERM + mt * BM + arow + i * 32];

    // B-tile loader mapping: 32 k-rows x 64 n-cols, float4 along n
    const int bk  = tid >> 4;                // 0..15 (x2 reps -> 32 k)
    const int bn4 = (tid & 15) * 4;
    const float* Wbase = weight + (size_t)e * D_IN * D_OUT;

    v8f acc[2][2] = {};

    // ---- prologue: fetch tile k0 = 0 into registers ----
    float4 aReg[4], bReg[2];
#pragma unroll
    for (int i = 0; i < 4; ++i) {
        aReg[i] = make_float4(0.f, 0.f, 0.f, 0.f);
        if (tokRow[i] >= 0)
            aReg[i] = *(const float4*)(tokens + (size_t)tokRow[i] * D_IN + ac4);
    }
#pragma unroll
    for (int i = 0; i < 2; ++i)
        bReg[i] = *(const float4*)(Wbase + (size_t)(bk + i * 16) * D_OUT + n0 + bn4);

    for (int k0 = 0; k0 < D_IN; k0 += BK) {
        // ---- stage current registers into LDS as hi/lo bf16 ----
#pragma unroll
        for (int i = 0; i < 4; ++i) {
            const int r = arow + i * 32;
            const float xs[4] = {aReg[i].x, aReg[i].y, aReg[i].z, aReg[i].w};
#pragma unroll
            for (int j = 0; j < 4; ++j) {
                const float x = xs[j];
                const __bf16 h = (__bf16)x;
                const __bf16 l = (__bf16)(x - (float)h);
                AsHi[r * LDK + ac4 + j] = h;
                AsLo[r * LDK + ac4 + j] = l;
            }
        }
#pragma unroll
        for (int i = 0; i < 2; ++i) {
            const int k = bk + i * 16;
            const float xs[4] = {bReg[i].x, bReg[i].y, bReg[i].z, bReg[i].w};
#pragma unroll
            for (int j = 0; j < 4; ++j) {
                const float x = xs[j];
                const __bf16 h = (__bf16)x;
                const __bf16 l = (__bf16)(x - (float)h);
                BsHi[(bn4 + j) * LDK + k] = h;
                BsLo[(bn4 + j) * LDK + k] = l;
            }
        }

        __syncthreads();

        // ---- issue global loads for tile k0+BK (overlap with WMMAs below) ----
        const int kn = k0 + BK;
        if (kn < D_IN) {
#pragma unroll
            for (int i = 0; i < 4; ++i) {
                aReg[i] = make_float4(0.f, 0.f, 0.f, 0.f);
                if (tokRow[i] >= 0)
                    aReg[i] = *(const float4*)(tokens + (size_t)tokRow[i] * D_IN + kn + ac4);
            }
#pragma unroll
            for (int i = 0; i < 2; ++i)
                bReg[i] = *(const float4*)(Wbase + (size_t)(kn + bk + i * 16) * D_OUT + n0 + bn4);
            // warm L2 two tiles ahead
            if (kn + BK < D_IN)
                __builtin_prefetch(Wbase + (size_t)(kn + BK + bk) * D_OUT + n0 + bn4, 0, 0);
        }

        // ---- load fragments (ISA 16-bit A layout: lane M=L%16, K-halves by L/16) ----
        const int khalf = lane >> 4;         // 0 or 1
        const int rm    = lane & 15;
        v16bf aHi[2], aLo[2], bHi[2], bLo[2];
#pragma unroll
        for (int mi = 0; mi < 2; ++mi) {
            const int row = wm * 32 + mi * 16 + rm;
            FragU uh, ul;
            uh.h[0] = *(const v8bf*)(&AsHi[row * LDK] + khalf * 8);
            uh.h[1] = *(const v8bf*)(&AsHi[row * LDK] + 16 + khalf * 8);
            ul.h[0] = *(const v8bf*)(&AsLo[row * LDK] + khalf * 8);
            ul.h[1] = *(const v8bf*)(&AsLo[row * LDK] + 16 + khalf * 8);
            aHi[mi] = uh.v; aLo[mi] = ul.v;
        }
#pragma unroll
        for (int ni = 0; ni < 2; ++ni) {
            const int col = wn * 32 + ni * 16 + rm;
            FragU uh, ul;
            uh.h[0] = *(const v8bf*)(&BsHi[col * LDK] + khalf * 8);
            uh.h[1] = *(const v8bf*)(&BsHi[col * LDK] + 16 + khalf * 8);
            ul.h[0] = *(const v8bf*)(&BsLo[col * LDK] + khalf * 8);
            ul.h[1] = *(const v8bf*)(&BsLo[col * LDK] + 16 + khalf * 8);
            bHi[ni] = uh.v; bLo[ni] = ul.v;
        }

        // ---- 3-term bf16 emulation of fp32 GEMM, fp32 accumulate ----
#pragma unroll
        for (int mi = 0; mi < 2; ++mi)
#pragma unroll
            for (int ni = 0; ni < 2; ++ni) {
                acc[mi][ni] = __builtin_amdgcn_wmma_f32_16x16x32_bf16(
                    false, aHi[mi], false, bHi[ni], (short)0, acc[mi][ni], false, false);
                acc[mi][ni] = __builtin_amdgcn_wmma_f32_16x16x32_bf16(
                    false, aHi[mi], false, bLo[ni], (short)0, acc[mi][ni], false, false);
                acc[mi][ni] = __builtin_amdgcn_wmma_f32_16x16x32_bf16(
                    false, aLo[mi], false, bHi[ni], (short)0, acc[mi][ni], false, false);
            }

        __syncthreads();
    }

    // ---- scatter results back to per-token rows ----
    // C layout: VGPR r holds (lanes 0-15: M=r, N=lane) / (lanes 16-31: M=8+r, N=lane-16)
    const int rhalf = (lane >> 4) * 8;
    const int cn    = lane & 15;
#pragma unroll
    for (int mi = 0; mi < 2; ++mi) {
#pragma unroll
        for (int ni = 0; ni < 2; ++ni) {
            const int colBase = n0 + wn * 32 + ni * 16 + cn;
#pragma unroll
            for (int r = 0; r < 8; ++r) {
                const int localRow = wm * 32 + mi * 16 + rhalf + r;
                const int tok = ws[WS_PERM + mt * BM + localRow];
                if (tok >= 0)
                    out[(size_t)tok * D_OUT + colBase] = acc[mi][ni][r];
            }
        }
    }
}

// ---------------- launch ----------------
extern "C" void kernel_launch(void* const* d_in, const int* in_sizes, int n_in,
                              void* d_out, int out_size, void* d_ws, size_t ws_size,
                              hipStream_t stream) {
    const float* tokens  = (const float*)d_in[0];
    const float* weight  = (const float*)d_in[1];
    const int*   exp_ids = (const int*)d_in[2];
    float*       out     = (float*)d_out;
    int*         ws      = (int*)d_ws;

    // build expert-sorted, 128-aligned token permutation
    moe_init_kernel<<<(WS_PERM + PERM_CAP + 255) / 256, 256, 0, stream>>>(ws);
    moe_count_kernel<<<T_TOK / 256, 256, 0, stream>>>(exp_ids, ws);
    moe_scan_kernel<<<1, 32, 0, stream>>>(ws);
    moe_scatter_kernel<<<T_TOK / 256, 256, 0, stream>>>(exp_ids, ws);

    // grouped GEMM over padded tiles
    dim3 grid(MT_MAX, D_OUT / BN);
    moe_gemm_kernel<<<grid, 256, 0, stream>>>(tokens, weight, ws, out);
}
